// SimpleCNN_ConvLSTM_27625229648586
// MI455X (gfx1250) — matrix operand order repaired and checked by reading, not compiled, not run on hardware
//
#include <hip/hip_runtime.h>
#include <hip/hip_bf16.h>
#include <math.h>

typedef __attribute__((ext_vector_type(16))) _Float16 v16h;
typedef __attribute__((ext_vector_type(8)))  _Float16 v8h;
typedef __attribute__((ext_vector_type(8)))  float    v8f;

union V16 { v16h v; v8h h[2]; };

#define WPB 4   // waves per block for WMMA kernels (128 threads)

// ---------------------------------------------------------------------------
// Weight packing: OIHW f32 -> WMMA A-fragment layout f16.
// Packed element order: [co_tile][kt][lane(32)][e16(16)], e16 = 2j+e indexes
// v16h element (VGPR j, half e).  A layout used here (16-bit A 16x32, contiguous k order, not the ISA table order):
//   k = kt*32 + (lane>>4)*16 + e16 ,  m = lane&15.
// K ordering: k = tap*Cin + ci  (tap = kh*3+kw), matching NHWC im2col.
// ---------------------------------------------------------------------------
__global__ void pack_weights(const float* __restrict__ w, _Float16* __restrict__ wp,
                             int Cout, int Cin, int taps, int total) {
  int idx = blockIdx.x * blockDim.x + threadIdx.x;
  if (idx >= total) return;
  int K  = taps * Cin;
  int KT = (K + 31) >> 5;
  int e16  = idx & 15;
  int lane = (idx >> 4) & 31;
  int kt   = (idx >> 9) % KT;
  int cot  = idx / (KT << 9);
  int half = lane >> 4, m = lane & 15;
  int j = e16 >> 1, e = e16 & 1;
  int k = kt * 32 + half * 16 + e16;
  int co = cot * 16 + m;
  float val = 0.0f;
  if (k < K) {
    int r = k / Cin, ci = k - r * Cin;
    val = w[(size_t)(co * Cin + ci) * taps + r];
  }
  wp[idx] = (_Float16)val;
}

// ---------------------------------------------------------------------------
// Implicit-GEMM 3x3 conv (pad=1), NHWC f16 in/out, f32 accumulate, bias+ReLU.
// UPS=1: input is half resolution (fused 2x nearest upsample).
// Register-blocked: one wave computes a 32(channel) x 32(pixel) tile
// (2 co-tiles x 2 pixel-tiles, 4 accumulators) -> 2 b128-load-pairs per WMMA.
// ---------------------------------------------------------------------------
template <int CIN, int UPS>
__global__ __launch_bounds__(32 * WPB)
void conv3x3_wmma(const _Float16* __restrict__ in, const _Float16* __restrict__ wp,
                  const float* __restrict__ bias, _Float16* __restrict__ out,
                  int H, int W, int Cout) {
  constexpr int KT = (9 * CIN) / 32;   // exact for CIN in {64,128,256}
  const int lane = threadIdx.x & 31;
  const int wave = threadIdx.x >> 5;
  const int half = lane >> 4;
  const int lm   = lane & 15;
  const int n    = blockIdx.z;
  const int cop  = blockIdx.y;                  // pair of co-tiles (32 channels)
  const int Npix = H * W;
  const int p0   = (blockIdx.x * WPB + wave) * 32;
  if (p0 >= Npix) return;                       // wave-uniform
  const int Hin = UPS ? (H >> 1) : H;
  const int Win = UPS ? (W >> 1) : W;

  int oh[2], ow[2];
#pragma unroll
  for (int nt = 0; nt < 2; ++nt) {
    int p = p0 + nt * 16 + lm;
    if (p >= Npix) p = Npix - 1;                // clamp (EXEC stays full)
    oh[nt] = p / W; ow[nt] = p - oh[nt] * W;
  }

  v8f acc[2][2] = {};
  const _Float16* wbase0 = wp + ((size_t)(2 * cop) * KT) * 512 + lane * 16;
  const _Float16* wbase1 = wbase0 + (size_t)KT * 512;

  for (int kt = 0; kt < KT; ++kt) {
    V16 a[2], b[2];
    a[0].h[0] = *(const v8h*)(wbase0 + (size_t)kt * 512);
    a[0].h[1] = *(const v8h*)(wbase0 + (size_t)kt * 512 + 8);
    a[1].h[0] = *(const v8h*)(wbase1 + (size_t)kt * 512);
    a[1].h[1] = *(const v8h*)(wbase1 + (size_t)kt * 512 + 8);

    int k0 = kt * 32 + half * 16;
    int r  = k0 / CIN;
    int ci0 = k0 - r * CIN;                 // always <= CIN-16: contiguous load
    int kh = r / 3, kw = r - kh * 3;
    v8h z = {};
#pragma unroll
    for (int nt = 0; nt < 2; ++nt) {
      int ih = oh[nt] + kh - 1, iw = ow[nt] + kw - 1;
      if (ih >= 0 && ih < H && iw >= 0 && iw < W) {
        if (UPS) { ih >>= 1; iw >>= 1; }
        const _Float16* src = in + (((size_t)n * Hin + ih) * Win + iw) * CIN + ci0;
        b[nt].h[0] = *(const v8h*)(src);
        b[nt].h[1] = *(const v8h*)(src + 8);
      } else { b[nt].h[0] = z; b[nt].h[1] = z; }
    }
#pragma unroll
    for (int mt = 0; mt < 2; ++mt)
#pragma unroll
      for (int nt = 0; nt < 2; ++nt)
        acc[mt][nt] = __builtin_amdgcn_wmma_f32_16x16x32_f16(
            false, a[mt].v, false, b[nt].v, (short)0, acc[mt][nt], false, false);
  }

#pragma unroll
  for (int nt = 0; nt < 2; ++nt) {
    if (p0 + nt * 16 + lm < Npix) {
      size_t pbase = (((size_t)n * H + oh[nt]) * W + ow[nt]) * Cout;
#pragma unroll
      for (int mt = 0; mt < 2; ++mt) {
        int co = cop * 32 + mt * 16 + half * 8;
        v8h o;
#pragma unroll
        for (int j = 0; j < 8; ++j) {
          float v = acc[mt][nt][j] + bias[co + j];
          o[j] = (_Float16)(v > 0.0f ? v : 0.0f);  // ReLU
        }
        *(v8h*)(out + pbase + co) = o;
      }
    }
  }
}

// ---------------------------------------------------------------------------
// Pointwise GEMM (LSTM gates): [P,512] f16 x packed[512,1024] -> [P,1024] f32+bias
// Same 32x32 register blocking.
// ---------------------------------------------------------------------------
__global__ __launch_bounds__(32 * WPB)
void gemm_wmma_pw(const _Float16* __restrict__ in, const _Float16* __restrict__ wp,
                  const float* __restrict__ bias, float* __restrict__ out, int P) {
  constexpr int CIN = 512, KT = 16, COUT = 1024;
  const int lane = threadIdx.x & 31;
  const int wave = threadIdx.x >> 5;
  const int half = lane >> 4;
  const int lm   = lane & 15;
  const int cop  = blockIdx.y;
  const int p0   = (blockIdx.x * WPB + wave) * 32;
  if (p0 >= P) return;

  int p[2];
#pragma unroll
  for (int nt = 0; nt < 2; ++nt) {
    p[nt] = p0 + nt * 16 + lm;
    if (p[nt] >= P) p[nt] = P - 1;
  }

  v8f acc[2][2] = {};
  const _Float16* wbase0 = wp + ((size_t)(2 * cop) * KT) * 512 + lane * 16;
  const _Float16* wbase1 = wbase0 + (size_t)KT * 512;
  const _Float16* src0 = in + (size_t)p[0] * CIN + half * 16;
  const _Float16* src1 = in + (size_t)p[1] * CIN + half * 16;

  for (int kt = 0; kt < KT; ++kt) {
    V16 a[2], b[2];
    a[0].h[0] = *(const v8h*)(wbase0 + (size_t)kt * 512);
    a[0].h[1] = *(const v8h*)(wbase0 + (size_t)kt * 512 + 8);
    a[1].h[0] = *(const v8h*)(wbase1 + (size_t)kt * 512);
    a[1].h[1] = *(const v8h*)(wbase1 + (size_t)kt * 512 + 8);
    b[0].h[0] = *(const v8h*)(src0 + kt * 32);
    b[0].h[1] = *(const v8h*)(src0 + kt * 32 + 8);
    b[1].h[0] = *(const v8h*)(src1 + kt * 32);
    b[1].h[1] = *(const v8h*)(src1 + kt * 32 + 8);
#pragma unroll
    for (int mt = 0; mt < 2; ++mt)
#pragma unroll
      for (int nt = 0; nt < 2; ++nt)
        acc[mt][nt] = __builtin_amdgcn_wmma_f32_16x16x32_f16(
            false, a[mt].v, false, b[nt].v, (short)0, acc[mt][nt], false, false);
  }

#pragma unroll
  for (int nt = 0; nt < 2; ++nt) {
    if (p0 + nt * 16 + lm < P) {
#pragma unroll
      for (int mt = 0; mt < 2; ++mt) {
        int co = cop * 32 + mt * 16 + half * 8;
        float* dst = out + (size_t)p[nt] * COUT + co;
#pragma unroll
        for (int j = 0; j < 8; ++j) dst[j] = acc[mt][nt][j] + bias[co + j];
      }
    }
  }
}

// ---------------------------------------------------------------------------
// Encoder conv1 (Cin=3) fused with ReLU + 2x2 maxpool.  x: f32 NCHW frames.
// out: [128,42,42,64] NHWC f16
// ---------------------------------------------------------------------------
__global__ void conv1_pool(const float* __restrict__ x, const float* __restrict__ w,
                           const float* __restrict__ bias, _Float16* __restrict__ out,
                           int total) {
  int idx = blockIdx.x * blockDim.x + threadIdx.x;
  if (idx >= total) return;
  int co = idx & 63;
  int t = idx >> 6;
  int ow = t % 42; t /= 42;
  int oh = t % 42; t /= 42;
  int n = t;
  float m = -1e30f;
#pragma unroll
  for (int dy = 0; dy < 2; ++dy)
#pragma unroll
    for (int dx = 0; dx < 2; ++dx) {
      int h = oh * 2 + dy, wp2 = ow * 2 + dx;
      float s = bias[co];
      for (int ci = 0; ci < 3; ++ci)
#pragma unroll
        for (int kh = 0; kh < 3; ++kh)
#pragma unroll
          for (int kw = 0; kw < 3; ++kw) {
            int ih = h + kh - 1, iw = wp2 + kw - 1;
            if (ih >= 0 && ih < 84 && iw >= 0 && iw < 84)
              s += x[(((size_t)n * 3 + ci) * 84 + ih) * 84 + iw] *
                   w[((co * 3 + ci) * 3 + kh) * 3 + kw];
          }
      m = fmaxf(m, s);
    }
  out[idx] = (_Float16)fmaxf(m, 0.0f);
}

// 2x2 maxpool, NHWC f16
__global__ void pool2(const _Float16* __restrict__ in, _Float16* __restrict__ out,
                      int Hi, int Wi, int C, int total) {
  int idx = blockIdx.x * blockDim.x + threadIdx.x;
  if (idx >= total) return;
  int Wo = Wi >> 1, Ho = Hi >> 1;
  int c = idx % C; int t = idx / C;
  int wo = t % Wo; t /= Wo;
  int ho = t % Ho; t /= Ho;
  int n = t;
  size_t base = (((size_t)n * Hi + ho * 2) * Wi + wo * 2) * C + c;
  float a = (float)in[base],          b = (float)in[base + C];
  float d = (float)in[base + (size_t)Wi * C], e = (float)in[base + (size_t)Wi * C + C];
  out[idx] = (_Float16)fmaxf(fmaxf(a, b), fmaxf(d, e));
}

// Depthwise 3x3 over concat(ft, h) -> g1 [8,21,21,512] f16
__global__ void dwconv(const _Float16* __restrict__ feats, const _Float16* __restrict__ hstate,
                       const float* __restrict__ dww, _Float16* __restrict__ g1,
                       int t, int total) {
  int idx = blockIdx.x * blockDim.x + threadIdx.x;
  if (idx >= total) return;
  int c = idx & 511; int t2 = idx >> 9;
  int x = t2 % 21; t2 /= 21;
  int y = t2 % 21; int b = t2 / 21;
  float s = 0.0f;
#pragma unroll
  for (int kh = 0; kh < 3; ++kh)
#pragma unroll
    for (int kw = 0; kw < 3; ++kw) {
      int ih = y + kh - 1, iw = x + kw - 1;
      if (ih >= 0 && ih < 21 && iw >= 0 && iw < 21) {
        int pix = ih * 21 + iw;
        float v = (c < 256)
          ? (float)feats[(((size_t)(b * 16 + t) * 441) + pix) * 256 + c]
          : (float)hstate[(((size_t)b * 441) + pix) * 256 + (c - 256)];
        s += v * dww[c * 9 + kh * 3 + kw];
      }
    }
  g1[idx] = (_Float16)s;
}

// LSTM gate nonlinearities + state update; also records h into hs[frame]
__global__ void lstm_elem(const float* __restrict__ g, float* __restrict__ cst,
                          _Float16* __restrict__ hst, _Float16* __restrict__ hs,
                          int t, int total) {
  int idx = blockIdx.x * blockDim.x + threadIdx.x;
  if (idx >= total) return;
  int c = idx & 255; int t2 = idx >> 8;
  int pix = t2 % 441; int b = t2 / 441;
  size_t gb = ((size_t)b * 441 + pix) * 1024;
  float gi = g[gb + c], gf = g[gb + 256 + c], go = g[gb + 512 + c], gg = g[gb + 768 + c];
  float si = 1.0f / (1.0f + expf(-gi));
  float sf = 1.0f / (1.0f + expf(-gf));
  float so = 1.0f / (1.0f + expf(-go));
  float c2 = sf * cst[idx] + si * tanhf(gg);
  float h2 = so * tanhf(c2);
  cst[idx] = c2;
  hst[idx] = (_Float16)h2;
  hs[(((size_t)(b * 16 + t) * 441) + pix) * 256 + c] = (_Float16)h2;
}

__global__ void init_state(_Float16* __restrict__ hst, float* __restrict__ cst, int total) {
  int idx = blockIdx.x * blockDim.x + threadIdx.x;
  if (idx >= total) return;
  hst[idx] = (_Float16)0.0f;
  cst[idx] = 0.0f;
}

// Final 1x1 conv 64->1 + bias, f32 out (the [8,16,1,84,84] image output)
__global__ void conv1x1_out(const _Float16* __restrict__ d2, const float* __restrict__ w,
                            const float* __restrict__ b, float* __restrict__ out, int total) {
  int idx = blockIdx.x * blockDim.x + threadIdx.x;
  if (idx >= total) return;
  const _Float16* src = d2 + (size_t)idx * 64;
  float s = b[0];
#pragma unroll
  for (int i = 0; i < 64; ++i) s += (float)src[i] * w[i];
  out[idx] = s;
}

// NHWC state -> NCHW f32 outputs for h, c
__global__ void finalize_hc(const _Float16* __restrict__ hst, const float* __restrict__ cst,
                            float* __restrict__ outh, float* __restrict__ outc, int total) {
  int idx = blockIdx.x * blockDim.x + threadIdx.x;
  if (idx >= total) return;
  int pix = idx % 441; int t2 = idx / 441;
  int ch = t2 % 256; int b = t2 / 256;
  size_t s = (((size_t)b * 441) + pix) * 256 + ch;
  outh[idx] = (float)hst[s];
  outc[idx] = cst[s];
}

// ---------------------------------------------------------------------------
extern "C" void kernel_launch(void* const* d_in, const int* in_sizes, int n_in,
                              void* d_out, int out_size, void* d_ws, size_t ws_size,
                              hipStream_t stream) {
  const float* x      = (const float*)d_in[0];
  const float* enc_w1 = (const float*)d_in[1];  const float* enc_b1 = (const float*)d_in[2];
  const float* enc_w2 = (const float*)d_in[3];  const float* enc_b2 = (const float*)d_in[4];
  const float* enc_w3 = (const float*)d_in[5];  const float* enc_b3 = (const float*)d_in[6];
  const float* dw_w   = (const float*)d_in[7];
  const float* pw_w   = (const float*)d_in[8];  const float* pw_b   = (const float*)d_in[9];
  const float* dec_w1 = (const float*)d_in[10]; const float* dec_b1 = (const float*)d_in[11];
  const float* dec_w2 = (const float*)d_in[12]; const float* dec_b2 = (const float*)d_in[13];
  const float* dec_w3 = (const float*)d_in[14]; const float* dec_b3 = (const float*)d_in[15];
  float* out_img = (float*)d_out;              // 8*16*1*84*84 = 903168
  float* out_h   = out_img + 903168;           // 8*256*21*21
  float* out_c   = out_h + 903168;

  // ---- workspace bump allocator (256B aligned) with lifetime aliasing ----
  size_t off = 0;
  auto alloc = [&](size_t bytes) -> void* {
    void* p = (char*)d_ws + off;
    off += (bytes + 255) & ~(size_t)255;
    return p;
  };
  // packed weights (persistent)
  _Float16* wp_e2 = (_Float16*)alloc(73728 * 2);    // 128x(9*64)
  _Float16* wp_e3 = (_Float16*)alloc(294912 * 2);   // 256x(9*128)
  _Float16* wp_pw = (_Float16*)alloc(524288 * 2);   // 1024x512
  _Float16* wp_d1 = (_Float16*)alloc(294912 * 2);   // 128x(9*256)
  _Float16* wp_d2 = (_Float16*)alloc(73728 * 2);    // 64x(9*128)
  // big aliased regions
  char* bufA = (char*)alloc(115605504);  // p1 [128,42,42,64]f16  then d2 [128,84,84,64]f16
  char* bufB = (char*)alloc(57802752);   // a2 [128,42,42,128]f16 then d1 [128,42,42,128]f16
  char* bufC = (char*)alloc(14450688);   // p2 [128,21,21,128]f16 then gates [8,441,1024]f32
  _Float16* feats  = (_Float16*)alloc(14450688ull * 2);  // [128,21,21,256]
  _Float16* hs     = (_Float16*)alloc(14450688ull * 2);  // [128,21,21,256]
  _Float16* g1     = (_Float16*)alloc(1806336ull * 2);   // [8,21,21,512]
  _Float16* hstate = (_Float16*)alloc(903168ull * 2);    // [8,441,256]
  float*    cstate = (float*)alloc(903168ull * 4);
  (void)ws_size; (void)in_sizes; (void)n_in; (void)out_size;

  _Float16* p1 = (_Float16*)bufA;  _Float16* d2b = (_Float16*)bufA;
  _Float16* a2 = (_Float16*)bufB;  _Float16* d1b = (_Float16*)bufB;
  _Float16* p2 = (_Float16*)bufC;  float*    gts = (float*)bufC;

  const int TB = 256;
  auto cdiv = [](int a, int b) { return (a + b - 1) / b; };

  // ---- pack weights into WMMA A-fragment layout ----
  pack_weights<<<cdiv(73728, TB),  TB, 0, stream>>>(enc_w2, wp_e2, 128, 64, 9, 73728);
  pack_weights<<<cdiv(294912, TB), TB, 0, stream>>>(enc_w3, wp_e3, 256, 128, 9, 294912);
  pack_weights<<<cdiv(524288, TB), TB, 0, stream>>>(pw_w,   wp_pw, 1024, 512, 1, 524288);
  pack_weights<<<cdiv(294912, TB), TB, 0, stream>>>(dec_w1, wp_d1, 128, 256, 9, 294912);
  pack_weights<<<cdiv(73728, TB),  TB, 0, stream>>>(dec_w2, wp_d2, 64, 128, 9, 73728);

  // ---- encoder ----
  conv1_pool<<<cdiv(14450688, TB), TB, 0, stream>>>(x, enc_w1, enc_b1, p1, 14450688);

  { // conv2: 64->128 @42x42 + ReLU   (1764 px -> 56 tiles of 32)
    dim3 g(cdiv(cdiv(1764, 32), WPB), 128 / 32, 128);
    conv3x3_wmma<64, 0><<<g, 32 * WPB, 0, stream>>>(p1, wp_e2, enc_b2, a2, 42, 42, 128);
  }
  pool2<<<cdiv(7225344, TB), TB, 0, stream>>>(a2, p2, 42, 42, 128, 7225344);
  { // conv3: 128->256 @21x21 + ReLU -> feats  (441 px -> 14 tiles of 32)
    dim3 g(cdiv(cdiv(441, 32), WPB), 256 / 32, 128);
    conv3x3_wmma<128, 0><<<g, 32 * WPB, 0, stream>>>(p2, wp_e3, enc_b3, feats, 21, 21, 256);
  }

  // ---- ConvLSTM: 16 steps ----
  init_state<<<cdiv(903168, TB), TB, 0, stream>>>(hstate, cstate, 903168);
  for (int t = 0; t < 16; ++t) {
    dwconv<<<cdiv(1806336, TB), TB, 0, stream>>>(feats, hstate, dw_w, g1, t, 1806336);
    { // pointwise 512->1024 over P = 8*441 = 3528 pixels (111 tiles of 32)
      dim3 g(cdiv(cdiv(3528, 32), WPB), 1024 / 32, 1);
      gemm_wmma_pw<<<g, 32 * WPB, 0, stream>>>(g1, wp_pw, pw_b, gts, 3528);
    }
    lstm_elem<<<cdiv(903168, TB), TB, 0, stream>>>(gts, cstate, hstate, hs, t, 903168);
  }

  // ---- decoder (2x upsample fused into implicit GEMM) ----
  { // up2 + conv 256->128 @42x42 + ReLU
    dim3 g(cdiv(cdiv(1764, 32), WPB), 128 / 32, 128);
    conv3x3_wmma<256, 1><<<g, 32 * WPB, 0, stream>>>(hs, wp_d1, dec_b1, d1b, 42, 42, 128);
  }
  { // up2 + conv 128->64 @84x84 + ReLU  (7056 px -> 221 tiles of 32)
    dim3 g(cdiv(cdiv(7056, 32), WPB), 64 / 32, 128);
    conv3x3_wmma<128, 1><<<g, 32 * WPB, 0, stream>>>(d1b, wp_d2, dec_b2, d2b, 84, 84, 64);
  }
  conv1x1_out<<<cdiv(903168, TB), TB, 0, stream>>>(d2b, dec_w3, dec_b3, out_img, 903168);

  // ---- final states h, c as NCHW f32 ----
  finalize_hc<<<cdiv(903168, TB), TB, 0, stream>>>(hstate, cstate, out_h, out_c, 903168);
}
